// MultiHeadAttention_4492535791776
// MI455X (gfx1250) — matrix operand order repaired and checked by reading, not compiled, not run on hardware
//
#include <hip/hip_runtime.h>
#include <stdint.h>

typedef __bf16 bf16_t;
typedef __attribute__((ext_vector_type(16))) __bf16 v16bf;
typedef __attribute__((ext_vector_type(8)))  __bf16 v8bf;
typedef __attribute__((ext_vector_type(8)))  float  v8f;
typedef __attribute__((ext_vector_type(4)))  uint32_t u32x4;
typedef __attribute__((ext_vector_type(8)))  uint32_t u32x8;

#define D_EMBED 2048
#define N_HEAD  16
#define HEAD_DIM 128
#define BATCH   4
#define SEQ     2048
#define ROWS    (BATCH*SEQ)   /* 8192 */
#define KSTEP   64

// Build a 16-element bf16 fragment from two contiguous 8-element (16B) chunks.
__device__ inline v16bf make_frag(const bf16_t* lo_p, const bf16_t* hi_p) {
  v8bf lo = *(const v8bf*)lo_p;
  v8bf hi = *(const v8bf*)hi_p;
  v16bf a;
#pragma unroll
  for (int i = 0; i < 8; ++i) { a[i] = lo[i]; a[8 + i] = hi[i]; }
  return a;
}

__device__ inline v8f wmma_bf16(v16bf a, v16bf b, v8f c) {
  return __builtin_amdgcn_wmma_f32_16x16x32_bf16(false, a, false, b, (short)0, c,
                                                 false, false);
}

__device__ inline float rmax16(float v) {
#pragma unroll
  for (int m = 1; m < 16; m <<= 1) v = fmaxf(v, __shfl_xor(v, m, 16));
  return v;
}
__device__ inline float rsum16(float v) {
#pragma unroll
  for (int m = 1; m < 16; m <<= 1) v += __shfl_xor(v, m, 16);
  return v;
}

// ---------------------------------------------------------------------------
// TDM descriptor helpers (CDNA5 ISA ch.8): 2-D tile copy, bf16 elements.
// Group0: count=1 | lds_addr[63:32] | global_addr[120:64] | type=2 (127:126)
// Group1: data_size=1(2B) @bits17:16 | tensor_dim0 @79:48 | tensor_dim1 @111:80
//         tile_dim0 @127:112 | tile_dim1 @143:128 | dim0_stride @207:160
// ---------------------------------------------------------------------------
__device__ inline u32x8 tdm_g1(uint32_t tdim0, uint32_t tdim1,
                               uint32_t tile0, uint32_t tile1,
                               uint64_t stride0) {
  u32x8 g;
  g[0] = 1u << 16;                                        // data_size = 2B
  g[1] = (tdim0 & 0xFFFFu) << 16;                         // tensor_dim0[15:0]
  g[2] = (tdim0 >> 16) | ((tdim1 & 0xFFFFu) << 16);       // dim0 hi | dim1 lo
  g[3] = (tdim1 >> 16) | ((tile0 & 0xFFFFu) << 16);       // dim1 hi | tile_dim0
  g[4] = tile1 & 0xFFFFu;                                 // tile_dim1; tile_dim2=0
  g[5] = (uint32_t)(stride0 & 0xFFFFFFFFu);               // dim0_stride lo
  g[6] = (uint32_t)((stride0 >> 32) & 0xFFFFu);           // dim0_stride hi
  g[7] = 0;                                               // dim1_stride = 0 (2D)
  return g;
}
__device__ inline u32x4 tdm_g0(uint32_t lds_addr, uint64_t gaddr) {
  u32x4 g;
  g[0] = 1u;                                              // count = 1
  g[1] = lds_addr;
  g[2] = (uint32_t)(gaddr & 0xFFFFFFFFu);
  g[3] = ((uint32_t)((gaddr >> 32) & 0x1FFFFFFu)) | (2u << 30);  // addr hi | type=2
  return g;
}
__device__ inline void tdm_load(u32x4 g0, u32x8 g1) {
  asm volatile("tensor_load_to_lds %0, %1" :: "s"(g0), "s"(g1) : "memory");
}
__device__ inline uint32_t lds_addr_of(const void* p) {
  return (uint32_t)(size_t)p;   // LDS flat addr: group-segment offset in [31:0]
}

// ---------------------------------------------------------------------------
// Kernel 0a: x (f32) -> xh (bf16), same layout [8192,2048]
// ---------------------------------------------------------------------------
__global__ __launch_bounds__(256)
void cvt_x_kernel(const float* __restrict__ x, bf16_t* __restrict__ xh) {
  size_t i = ((size_t)blockIdx.x * 256 + threadIdx.x) * 4;
  float4 v = *(const float4*)(x + i);
  xh[i + 0] = (bf16_t)v.x;
  xh[i + 1] = (bf16_t)v.y;
  xh[i + 2] = (bf16_t)v.z;
  xh[i + 3] = (bf16_t)v.w;
}

// ---------------------------------------------------------------------------
// Kernel 0b: W[z] (f32 [k][n]) -> WT[z] (bf16 [n][k]) transposed via LDS tile
// ---------------------------------------------------------------------------
__global__ __launch_bounds__(256)
void cvt_wT_kernel(const float* __restrict__ Wq, const float* __restrict__ Wk,
                   const float* __restrict__ Wv, const float* __restrict__ Wo,
                   bf16_t* __restrict__ WT) {
  const int z = blockIdx.z;
  const float* W = (z == 0) ? Wq : (z == 1) ? Wk : (z == 2) ? Wv : Wo;
  bf16_t* dst = WT + (size_t)z * D_EMBED * D_EMBED;
  __shared__ float tile[32][33];
  const int tx = threadIdx.x & 31, ty = threadIdx.x >> 5;  // 32 x 8
  const int n0 = blockIdx.x * 32, k0 = blockIdx.y * 32;
#pragma unroll
  for (int j = 0; j < 32; j += 8)
    tile[ty + j][tx] = W[(size_t)(k0 + ty + j) * D_EMBED + n0 + tx];
  __syncthreads();
#pragma unroll
  for (int j = 0; j < 32; j += 8)
    dst[(size_t)(n0 + ty + j) * D_EMBED + k0 + tx] = (bf16_t)tile[tx][ty + j];
}

// ---------------------------------------------------------------------------
// TDM-fed GEMM core: C[128x128] = A[bf16, M x 2048] x WT[bf16, [n][k]]^T
// K-step 64, double-buffered LDS, wave0 drives the Tensor Data Mover.
// 8 waves in 4(m) x 2(n); each wave: 32x64 output = 8 accumulators.
// ---------------------------------------------------------------------------
__device__ inline void gemm_tdm_core(const bf16_t* __restrict__ A, int a_rows,
                                     const bf16_t* __restrict__ WT,
                                     bf16_t (*sA)[128 * KSTEP],
                                     bf16_t (*sB)[128 * KSTEP],
                                     int m0, int n0, int wid, int lane,
                                     v8f acc[2][4]) {
  const int hi = lane >> 4, ln = lane & 15;
  const int wm = wid & 3, wn = wid >> 2;

  const u32x8 g1A = tdm_g1(D_EMBED, a_rows, KSTEP, 128, D_EMBED);
  const u32x8 g1B = tdm_g1(D_EMBED, D_EMBED, KSTEP, 128, D_EMBED);
  const uint64_t gA0 = (uint64_t)(size_t)A + ((uint64_t)m0 * D_EMBED) * 2;
  const uint64_t gB0 = (uint64_t)(size_t)WT + ((uint64_t)n0 * D_EMBED) * 2;
  const uint32_t ldsA[2] = {lds_addr_of(&sA[0][0]), lds_addr_of(&sA[1][0])};
  const uint32_t ldsB[2] = {lds_addr_of(&sB[0][0]), lds_addr_of(&sB[1][0])};

  const int iters = D_EMBED / KSTEP;  // 32

  if (wid == 0) {
    tdm_load(tdm_g0(ldsA[0], gA0), g1A);
    tdm_load(tdm_g0(ldsB[0], gB0), g1B);
  }

  int buf = 0;
  for (int it = 0; it < iters; ++it) {
    if (wid == 0) {
      if (it + 1 < iters) {
        uint64_t off = (uint64_t)(it + 1) * KSTEP * 2;
        tdm_load(tdm_g0(ldsA[buf ^ 1], gA0 + off), g1A);
        tdm_load(tdm_g0(ldsB[buf ^ 1], gB0 + off), g1B);
        asm volatile("s_wait_tensorcnt 0x2" ::: "memory");
      } else {
        asm volatile("s_wait_tensorcnt 0x0" ::: "memory");
      }
    }
    __syncthreads();

    const bf16_t* a_base = &sA[buf][0];
    const bf16_t* b_base = &sB[buf][0];
#pragma unroll
    for (int c = 0; c < 2; ++c) {
      v16bf afr[2], bfr[4];
#pragma unroll
      for (int mi = 0; mi < 2; ++mi) {
        const bf16_t* p = a_base + (wm * 32 + mi * 16 + ln) * KSTEP + c * 32;
        afr[mi] = make_frag(p + hi * 16, p + hi * 16 + 8);
      }
#pragma unroll
      for (int ni = 0; ni < 4; ++ni) {
        const bf16_t* p =
            b_base + (wn * 64 + ni * 16 + ln) * KSTEP + c * 32 + hi * 16;
        bfr[ni] = make_frag(p, p + 8);
      }
#pragma unroll
      for (int mi = 0; mi < 2; ++mi)
#pragma unroll
        for (int ni = 0; ni < 4; ++ni)
          acc[mi][ni] = wmma_bf16(afr[mi], bfr[ni], acc[mi][ni]);
    }
    __syncthreads();
    buf ^= 1;
  }
}

// ---------------------------------------------------------------------------
// Kernel 1: QKV projections via TDM GEMM; epilogue remaps to
//   Q,K: [B,H,S,hd]   V: [B,H,hd,S] (transposed), bf16.
// ---------------------------------------------------------------------------
__global__ __launch_bounds__(256)
void qkv_gemm_kernel(const bf16_t* __restrict__ xh, const bf16_t* __restrict__ WT,
                     const float* __restrict__ bq, const float* __restrict__ bk,
                     const float* __restrict__ bv,
                     bf16_t* __restrict__ Qh, bf16_t* __restrict__ Kh,
                     bf16_t* __restrict__ Vt) {
  __shared__ __align__(16) bf16_t sA[2][128 * KSTEP];
  __shared__ __align__(16) bf16_t sB[2][128 * KSTEP];

  const int z = blockIdx.z;
  const bf16_t* Wz = WT + (size_t)z * D_EMBED * D_EMBED;
  const float* bias = (z == 0) ? bq : (z == 1) ? bk : bv;

  const int tid = threadIdx.x;
  const int lane = tid & 31, wid = tid >> 5;
  const int hi = lane >> 4, ln = lane & 15;
  const int wm = wid & 3, wn = wid >> 2;
  const int m0 = blockIdx.y * 128, n0 = blockIdx.x * 128;

  v8f acc[2][4];
#pragma unroll
  for (int mi = 0; mi < 2; ++mi)
#pragma unroll
    for (int ni = 0; ni < 4; ++ni) acc[mi][ni] = (v8f){0,0,0,0,0,0,0,0};

  gemm_tdm_core(xh, ROWS, Wz, sA, sB, m0, n0, wid, lane, acc);

#pragma unroll
  for (int mi = 0; mi < 2; ++mi)
#pragma unroll
    for (int ni = 0; ni < 4; ++ni) {
      int col = n0 + wn * 64 + ni * 16 + ln;
      float bv_ = bias[col];
      int h = col >> 7, d = col & 127;
#pragma unroll
      for (int r = 0; r < 8; ++r) {
        int row = m0 + wm * 32 + mi * 16 + hi * 8 + r;
        int b = row >> 11, s = row & 2047;
        float val = acc[mi][ni][r] + bv_;
        if (z == 0)
          Qh[(((size_t)(b * N_HEAD + h) * SEQ + s) << 7) + d] = (bf16_t)val;
        else if (z == 1)
          Kh[(((size_t)(b * N_HEAD + h) * SEQ + s) << 7) + d] = (bf16_t)val;
        else
          Vt[((size_t)(b * N_HEAD + h) * HEAD_DIM + d) * SEQ + s] = (bf16_t)val;
      }
    }
}

// ---------------------------------------------------------------------------
// Kernel 2: flash attention (unchanged structure). One wave per
// (b,h,16-query-rows); K/V stream from L2 (whole bf16 K+V = 64MB, L2-resident).
// ---------------------------------------------------------------------------
__global__ __launch_bounds__(256)
void flash_attn_kernel(const bf16_t* __restrict__ Qh,
                       const bf16_t* __restrict__ Kh,
                       const bf16_t* __restrict__ Vt,
                       const uint8_t* __restrict__ am,
                       bf16_t* __restrict__ ctx) {
  __shared__ __align__(16) bf16_t pbuf[8][16 * 32];

  const int tid = threadIdx.x;
  const int lane = tid & 31, wid = tid >> 5;
  const int hi = lane >> 4, ln = lane & 15;
  const int id = blockIdx.x * 8 + wid;
  const int qt = id & (SEQ / 16 - 1);
  const int bh = id >> 7;
  const int b = bh >> 4, h = bh & 15;
  const int qbase = qt * 16;

  const bf16_t* Qb = Qh + (size_t)bh * SEQ * HEAD_DIM;
  const bf16_t* Kb = Kh + (size_t)bh * SEQ * HEAD_DIM;
  const bf16_t* Vb = Vt + (size_t)bh * HEAD_DIM * SEQ;
  const uint8_t* amb = am + (size_t)b * SEQ;
  const float scale = 0.088388347648318447f;  // 1/sqrt(128)

  v16bf qfr[4];
  {
    const bf16_t* qrow = Qb + (size_t)(qbase + ln) * HEAD_DIM;
#pragma unroll
    for (int c = 0; c < 4; ++c)
      qfr[c] = make_frag(qrow + c * 32 + hi * 16, qrow + c * 32 + hi * 16 + 8);
  }

  v8f acc[8];
#pragma unroll
  for (int t = 0; t < 8; ++t) acc[t] = (v8f){0,0,0,0,0,0,0,0};
  float mrow[8], lrow[8];
#pragma unroll
  for (int r = 0; r < 8; ++r) { mrow[r] = -1e30f; lrow[r] = 0.0f; }

  bf16_t* pb = pbuf[wid];

  for (int kt = 0; kt < qbase + 16; kt += 32) {
    v8f c0 = (v8f){0,0,0,0,0,0,0,0};
    v8f c1 = (v8f){0,0,0,0,0,0,0,0};
#pragma unroll
    for (int c = 0; c < 4; ++c) {
      const bf16_t* k0p = Kb + (size_t)(kt + ln) * HEAD_DIM + c * 32 + hi * 16;
      const bf16_t* k1p = Kb + (size_t)(kt + 16 + ln) * HEAD_DIM + c * 32 + hi * 16;
      v16bf b0 = make_frag(k0p, k0p + 8);
      v16bf b1 = make_frag(k1p, k1p + 8);
      c0 = wmma_bf16(qfr[c], b0, c0);
      c1 = wmma_bf16(qfr[c], b1, c1);
    }

    const int col0 = kt + ln, col1 = kt + 16 + ln;
    const bool ok0 = amb[col0] != 0;
    const bool ok1 = amb[col1] != 0;

#pragma unroll
    for (int r = 0; r < 8; ++r) {
      const int qg = qbase + hi * 8 + r;
      float s0 = (ok0 && col0 <= qg) ? c0[r] * scale : -1e30f;
      float s1 = (ok1 && col1 <= qg) ? c1[r] * scale : -1e30f;
      float tm = rmax16(fmaxf(s0, s1));
      float newm = fmaxf(mrow[r], tm);
      float alpha = __expf(mrow[r] - newm);
      mrow[r] = newm;
      float p0 = __expf(s0 - newm);
      float p1 = __expf(s1 - newm);
      lrow[r] = lrow[r] * alpha + rsum16(p0 + p1);
#pragma unroll
      for (int t = 0; t < 8; ++t) acc[t][r] *= alpha;
      pb[(hi * 8 + r) * 32 + ln] = (bf16_t)p0;
      pb[(hi * 8 + r) * 32 + 16 + ln] = (bf16_t)p1;
    }

    asm volatile("s_wait_dscnt 0x0" ::: "memory");

    const bf16_t* prow = &pb[ln * 32];
    v16bf pf = make_frag(prow + hi * 16, prow + hi * 16 + 8);
#pragma unroll
    for (int t = 0; t < 8; ++t) {
      const bf16_t* vp = Vb + (size_t)(t * 16 + ln) * SEQ + kt + hi * 16;
      v16bf vf = make_frag(vp, vp + 8);
      acc[t] = wmma_bf16(pf, vf, acc[t]);
    }
    asm volatile("s_wait_dscnt 0x0" ::: "memory");
  }

#pragma unroll
  for (int r = 0; r < 8; ++r) {
    float inv = 1.0f / lrow[r];
    int s = qbase + hi * 8 + r;
#pragma unroll
    for (int t = 0; t < 8; ++t)
      ctx[((size_t)(b * SEQ + s)) * D_EMBED + h * HEAD_DIM + t * 16 + ln] =
          (bf16_t)(acc[t][r] * inv);
  }
}

// ---------------------------------------------------------------------------
// Kernel 3: output projection via TDM GEMM; f32 output + bias.
// ---------------------------------------------------------------------------
__global__ __launch_bounds__(256)
void out_proj_kernel(const bf16_t* __restrict__ ctx, const bf16_t* __restrict__ WoT,
                     const float* __restrict__ bo, float* __restrict__ out) {
  __shared__ __align__(16) bf16_t sA[2][128 * KSTEP];
  __shared__ __align__(16) bf16_t sB[2][128 * KSTEP];

  const int tid = threadIdx.x;
  const int lane = tid & 31, wid = tid >> 5;
  const int hi = lane >> 4, ln = lane & 15;
  const int wm = wid & 3, wn = wid >> 2;
  const int m0 = blockIdx.y * 128, n0 = blockIdx.x * 128;

  v8f acc[2][4];
#pragma unroll
  for (int mi = 0; mi < 2; ++mi)
#pragma unroll
    for (int ni = 0; ni < 4; ++ni) acc[mi][ni] = (v8f){0,0,0,0,0,0,0,0};

  gemm_tdm_core(ctx, ROWS, WoT, sA, sB, m0, n0, wid, lane, acc);

#pragma unroll
  for (int mi = 0; mi < 2; ++mi)
#pragma unroll
    for (int ni = 0; ni < 4; ++ni) {
      int col = n0 + wn * 64 + ni * 16 + ln;
      float bv_ = bo[col];
#pragma unroll
      for (int r = 0; r < 8; ++r) {
        int row = m0 + wm * 32 + mi * 16 + hi * 8 + r;
        out[(size_t)row * D_EMBED + col] = acc[mi][ni][r] + bv_;
      }
    }
}

// ---------------------------------------------------------------------------
// Workspace layout (160 MB total):
//   [  0, 32) MB : xh (bf16 x)  -- reused as ctx after QKV GEMM completes
//   [ 32, 64) MB : WT  (4 x bf16 transposed weights, 8MB each: q,k,v,o)
//   [ 64, 96) MB : Qh
//   [ 96,128) MB : Kh
//   [128,160) MB : Vt (transposed V)
// ---------------------------------------------------------------------------
extern "C" void kernel_launch(void* const* d_in, const int* in_sizes, int n_in,
                              void* d_out, int out_size, void* d_ws,
                              size_t ws_size, hipStream_t stream) {
  const float*   x  = (const float*)d_in[0];
  const uint8_t* am = (const uint8_t*)d_in[1];
  const float* Wq = (const float*)d_in[2];
  const float* bq = (const float*)d_in[3];
  const float* Wk = (const float*)d_in[4];
  const float* bk = (const float*)d_in[5];
  const float* Wv = (const float*)d_in[6];
  const float* bv = (const float*)d_in[7];
  const float* Wo = (const float*)d_in[8];
  const float* bo = (const float*)d_in[9];
  float* out = (float*)d_out;

  char* ws = (char*)d_ws;
  bf16_t* xh  = (bf16_t*)(ws);                        // 32 MB (aliased by ctx)
  bf16_t* WT  = (bf16_t*)(ws + ((size_t)32 << 20));   // 32 MB (4 weights)
  bf16_t* Qh  = (bf16_t*)(ws + ((size_t)64 << 20));   // 32 MB
  bf16_t* Kh  = (bf16_t*)(ws + ((size_t)96 << 20));   // 32 MB
  bf16_t* Vt  = (bf16_t*)(ws + ((size_t)128 << 20));  // 32 MB
  bf16_t* ctx = xh;  // x is consumed by QKV GEMM before flash writes ctx

  dim3 blk(256);
  cvt_x_kernel<<<dim3(ROWS * D_EMBED / 4 / 256), blk, 0, stream>>>(x, xh);
  cvt_wT_kernel<<<dim3(D_EMBED / 32, D_EMBED / 32, 4), blk, 0, stream>>>(
      Wq, Wk, Wv, Wo, WT);

  qkv_gemm_kernel<<<dim3(D_EMBED / 128, ROWS / 128, 3), blk, 0, stream>>>(
      xh, WT, bq, bk, bv, Qh, Kh, Vt);

  flash_attn_kernel<<<dim3((BATCH * N_HEAD * (SEQ / 16)) / 8), blk, 0, stream>>>(
      Qh, Kh, Vt, am, ctx);

  out_proj_kernel<<<dim3(D_EMBED / 128, ROWS / 128), blk, 0, stream>>>(
      ctx, WT + (size_t)3 * D_EMBED * D_EMBED, bo, out);
}
